// MultiHeadAttention_70832600646193
// MI455X (gfx1250) — matrix operand order repaired and checked by reading, not compiled, not run on hardware
//
#include <hip/hip_runtime.h>
#include <hip/hip_bf16.h>

typedef __bf16 bf16;
typedef bf16  v16bf __attribute__((ext_vector_type(16)));
typedef bf16  v8bf  __attribute__((ext_vector_type(8)));
typedef float v8f   __attribute__((ext_vector_type(8)));
typedef unsigned int u32x4 __attribute__((ext_vector_type(4)));
typedef int  i32x4 __attribute__((ext_vector_type(4)));
typedef int  i32x8 __attribute__((ext_vector_type(8)));

// Fixed problem shape (reference: B=2, S=2048, D=1024, H=16, dk=64).
constexpr int CB = 2, CS = 2048, CD = 1024, CH = 16, CDK = 64;
constexpr int CM = CB * CS;   // 4096

__device__ __forceinline__ v8f vzero() {
    v8f z = {0.f, 0.f, 0.f, 0.f, 0.f, 0.f, 0.f, 0.f};
    return z;
}

// A-fragment (16x32 bf16, lane = M row): element j holds K = (j/8)*16 + half*8 + j%8.
__device__ __forceinline__ v16bf load_frag_a(const bf16* rowptr, int half) {
    v8bf lo = *(const v8bf*)(rowptr + half * 8);
    v8bf hi = *(const v8bf*)(rowptr + 16 + half * 8);
    return __builtin_shufflevector(lo, hi, 0,1,2,3,4,5,6,7,8,9,10,11,12,13,14,15);
}
// B-fragment (32x16 bf16, lane = N col): element j holds K = (j/8)*16 + half*8 + j%8.
__device__ __forceinline__ v16bf load_frag_b(const bf16* rowptr, int half) {
    v8bf lo = *(const v8bf*)(rowptr + half * 8);
    v8bf hi = *(const v8bf*)(rowptr + 16 + half * 8);
    return __builtin_shufflevector(lo, hi, 0,1,2,3,4,5,6,7,8,9,10,11,12,13,14,15);
}

__device__ __forceinline__ float rowmax16(float v) {
    v = fmaxf(v, __shfl_xor(v, 1, 32));
    v = fmaxf(v, __shfl_xor(v, 2, 32));
    v = fmaxf(v, __shfl_xor(v, 4, 32));
    v = fmaxf(v, __shfl_xor(v, 8, 32));
    return v;
}
__device__ __forceinline__ float rowsum16(float v) {
    v += __shfl_xor(v, 1, 32);
    v += __shfl_xor(v, 2, 32);
    v += __shfl_xor(v, 4, 32);
    v += __shfl_xor(v, 8, 32);
    return v;
}

// ---------------------------------------------------------------------------
// TDM: async 2D tile copy global -> LDS (D# per CDNA5 ISA).
// 2D tensor, data_size = 2 bytes, no padding/iterate/multicast.
// ---------------------------------------------------------------------------
__device__ __forceinline__ void tdm_load_tile(const bf16* gsrc, unsigned ldsByteAddr,
                                              int tileK, int tileRows,
                                              int strideElems, int dim0, int dim1) {
    unsigned long long ga = (unsigned long long)(size_t)gsrc;
    u32x4 g0;
    g0.x = 1u;                                    // count=1, user descriptor
    g0.y = ldsByteAddr;                           // lds_addr
    g0.z = (unsigned)ga;                          // global_addr[31:0]
    g0.w = (unsigned)(ga >> 32) | (2u << 30);     // global_addr[56:32] | type=2
    i32x8 g1;
    g1[0] = 1 << 16;                              // data_size=1 (2B); mask/flags=0
    g1[1] = (int)(((unsigned)dim0 & 0xFFFFu) << 16);          // tensor_dim0 lo16
    g1[2] = (int)(((unsigned)dim0 >> 16) |
                  (((unsigned)dim1 & 0xFFFFu) << 16));        // dim0 hi | dim1 lo
    g1[3] = (int)((((unsigned)dim1 >> 16) & 0xFFFFu) |
                  (((unsigned)tileK & 0xFFFFu) << 16));       // dim1 hi | tile_dim0
    g1[4] = (int)((unsigned)tileRows & 0xFFFFu);              // tile_dim1; tile_dim2=0
    g1[5] = strideElems;                                      // tensor_dim0_stride lo32
    g1[6] = 0;
    g1[7] = 0;
    i32x4 z4 = {0, 0, 0, 0};
#if defined(__clang_major__) && __clang_major__ >= 23
    i32x8 z8 = {0, 0, 0, 0, 0, 0, 0, 0};
    __builtin_amdgcn_tensor_load_to_lds(g0, g1, z4, z4, z8, 0);
#else
    __builtin_amdgcn_tensor_load_to_lds(g0, g1, z4, z4, 0);
#endif
}

// ---------------------------------------------------------------------------
// fp32 -> bf16 one-pass conversion (memory bound; all operands fit in 192MB L2).
// ---------------------------------------------------------------------------
__global__ void __launch_bounds__(256)
cvt_f32_bf16(const float* __restrict__ src, bf16* __restrict__ dst, int n8) {
    const int i = blockIdx.x * 256 + threadIdx.x;
    if (i < n8) {
        const float4* s4 = (const float4*)src;
        float4 a = s4[(size_t)i * 2];
        float4 b = s4[(size_t)i * 2 + 1];
        v8bf o;
        o[0] = (bf16)a.x; o[1] = (bf16)a.y; o[2] = (bf16)a.z; o[3] = (bf16)a.w;
        o[4] = (bf16)b.x; o[5] = (bf16)b.y; o[6] = (bf16)b.z; o[7] = (bf16)b.w;
        *(v8bf*)(dst + (size_t)i * 8) = o;
    }
}

// ---------------------------------------------------------------------------
// All-bf16 GEMM: Y = (A @ W^T + b) * oscale.  A: bf16 [4096,1024]; W: bf16
// [1024,1024]. Tiles staged via TDM, double-buffered.
// OUT_MODE 0: fp32 [M,N];  1: bf16 [B,H,S,dk] (Q,K);  2: bf16 [B,H,dk,S] (V).
// Block: 128 threads = 4 waves (2x2), tile 64x64, K-step 32. All dims constexpr.
// ---------------------------------------------------------------------------
template <int OUT_MODE>
__global__ void __launch_bounds__(128)
gemm_wmma(const bf16* __restrict__ A, const bf16* __restrict__ W,
          const float* __restrict__ bias, void* __restrict__ Out, float oscale) {
    constexpr int K = CD, N = CD;
    __shared__ bf16 ldsA[2][64 * 32];
    __shared__ bf16 ldsB[2][64 * 32];
    constexpr unsigned TILE_BYTES = 64 * 32 * 2;

    const int tid  = threadIdx.x;
    const int wave = tid >> 5;
    const int lane = tid & 31;
    const int half = lane >> 4;
    const int lr   = lane & 15;
    const int wm   = (wave >> 1) * 32;
    const int wn   = (wave & 1) * 32;
    const int m0   = blockIdx.x * 64;
    const int n0   = blockIdx.y * 64;

    const bf16* Arow = A + (size_t)m0 * K;
    const bf16* Wrow = W + (size_t)n0 * K;
    const unsigned ldsAaddr = (unsigned)(size_t)&ldsA[0][0];
    const unsigned ldsBaddr = (unsigned)(size_t)&ldsB[0][0];
    const bool issuer = (wave == 0);

    v8f acc[2][2];
#pragma unroll
    for (int i = 0; i < 2; ++i)
#pragma unroll
        for (int j = 0; j < 2; ++j) acc[i][j] = vzero();

    if (issuer) {
        tdm_load_tile(Arow, ldsAaddr, 32, 64, K, K, CM);
        tdm_load_tile(Wrow, ldsBaddr, 32, 64, K, K, N);
    }

    int cur = 0;
    for (int kk = 0; kk < K; kk += 32, cur ^= 1) {
        if (issuer) {
            if (kk + 32 < K) {
                tdm_load_tile(Arow + kk + 32, ldsAaddr + (cur ^ 1) * TILE_BYTES, 32, 64, K, K, CM);
                tdm_load_tile(Wrow + kk + 32, ldsBaddr + (cur ^ 1) * TILE_BYTES, 32, 64, K, K, N);
                __builtin_amdgcn_s_wait_tensorcnt(2);   // current pair landed
            } else {
                __builtin_amdgcn_s_wait_tensorcnt(0);
            }
        }
        __syncthreads();   // current tiles resident for all waves

        const bf16* lA = ldsA[cur];
        const bf16* lB = ldsB[cur];
        v16bf af0 = load_frag_a(lA + (wm + lr) * 32, half);
        v16bf af1 = load_frag_a(lA + (wm + 16 + lr) * 32, half);
        v16bf bf0 = load_frag_b(lB + (wn + lr) * 32, half);
        v16bf bf1 = load_frag_b(lB + (wn + 16 + lr) * 32, half);
        acc[0][0] = __builtin_amdgcn_wmma_f32_16x16x32_bf16(false, af0, false, bf0, (short)0, acc[0][0], false, false);
        acc[0][1] = __builtin_amdgcn_wmma_f32_16x16x32_bf16(false, af0, false, bf1, (short)0, acc[0][1], false, false);
        acc[1][0] = __builtin_amdgcn_wmma_f32_16x16x32_bf16(false, af1, false, bf0, (short)0, acc[1][0], false, false);
        acc[1][1] = __builtin_amdgcn_wmma_f32_16x16x32_bf16(false, af1, false, bf1, (short)0, acc[1][1], false, false);

        __syncthreads();   // everyone done reading before buffer is overwritten
    }

#pragma unroll
    for (int i = 0; i < 2; ++i) {
#pragma unroll
        for (int j = 0; j < 2; ++j) {
            const int col = n0 + wn + j * 16 + lr;
            const float bv = bias[col];
#pragma unroll
            for (int e = 0; e < 8; ++e) {
                const int row = m0 + wm + i * 16 + e + 8 * half;
                const float v = (acc[i][j][e] + bv) * oscale;
                if constexpr (OUT_MODE == 0) {
                    ((float*)Out)[(size_t)row * N + col] = v;
                } else {
                    const int bb = row >> 11, ss = row & (CS - 1);     // CS = 2048
                    const int hh = col >> 6,  dd = col & 63;
                    if constexpr (OUT_MODE == 1)
                        ((bf16*)Out)[(((size_t)bb * CH + hh) * CS + ss) * CDK + dd] = (bf16)v;
                    else
                        ((bf16*)Out)[(((size_t)bb * CH + hh) * CDK + dd) * CS + ss] = (bf16)v;
                }
            }
        }
    }
}

// ---------------------------------------------------------------------------
// One 64-wide key block of causal flash attention for one wave (16 q rows).
// MASKED=false for fully-unmasked blocks (kb + 63 < q for all rows).
// Q is pre-scaled by 1/sqrt(dk) in the projection, so no scale here.
// ---------------------------------------------------------------------------
template <bool MASKED>
__device__ __forceinline__ void attn_block(
    int kb, int q0w, int half, int lr,
    const bf16* __restrict__ Kbase, const bf16* __restrict__ Vbase, bf16* pw,
    const v16bf& aq0, const v16bf& aq1,
    v8f (&aco)[4], float (&mrow)[8], float (&lrow)[8]) {

    // ---- S = Q K^T (16 x 64 score tile) ----
    v8f acs[4];
#pragma unroll
    for (int t = 0; t < 4; ++t) acs[t] = vzero();
    const bf16* kr = Kbase + (size_t)(kb + lr) * CDK;
    {
        v16bf bk[4];
#pragma unroll
        for (int nt = 0; nt < 4; ++nt) bk[nt] = load_frag_b(kr + nt * 16 * CDK, half);
#pragma unroll
        for (int nt = 0; nt < 4; ++nt)
            acs[nt] = __builtin_amdgcn_wmma_f32_16x16x32_bf16(
                false, aq0, false, bk[nt], (short)0, acs[nt], false, false);
#pragma unroll
        for (int nt = 0; nt < 4; ++nt) bk[nt] = load_frag_b(kr + nt * 16 * CDK + 32, half);
#pragma unroll
        for (int nt = 0; nt < 4; ++nt)
            acs[nt] = __builtin_amdgcn_wmma_f32_16x16x32_bf16(
                false, aq1, false, bk[nt], (short)0, acs[nt], false, false);
    }

    // ---- online softmax (rows replicated across the 16-lane half) ----
#pragma unroll
    for (int e = 0; e < 8; ++e) {
        float s[4];
#pragma unroll
        for (int nt = 0; nt < 4; ++nt) {
            float v = acs[nt][e];
            if constexpr (MASKED) {
                const int q = q0w + e + 8 * half;
                const int kcol = kb + nt * 16 + lr;
                v = (kcol <= q) ? v : -1e30f;   // causal mask via select (EXEC stays full)
            }
            s[nt] = v;
        }
        float rm = fmaxf(fmaxf(s[0], s[1]), fmaxf(s[2], s[3]));
        rm = rowmax16(rm);
        const float mnew = fmaxf(mrow[e], rm);
        const float corr = __expf(mrow[e] - mnew);
        float psum = 0.f;
#pragma unroll
        for (int nt = 0; nt < 4; ++nt) {
            const float p = __expf(s[nt] - mnew);
            psum += p;
            pw[(e + 8 * half) * 64 + nt * 16 + lr] = (bf16)p;  // C-layout -> LDS
        }
        psum = rowsum16(psum);
        lrow[e] = lrow[e] * corr + psum;
        mrow[e] = mnew;
#pragma unroll
        for (int t = 0; t < 4; ++t) aco[t][e] *= corr;
    }

    // ---- O += P @ V (P re-read from per-wave LDS as A-fragments) ----
    const bf16* vr = Vbase + (size_t)lr * CS + kb;
    v16bf ap0 = load_frag_a(pw + lr * 64, half);
    v16bf ap1 = load_frag_a(pw + lr * 64 + 32, half);
    {
        v16bf bv[4];
#pragma unroll
        for (int nt = 0; nt < 4; ++nt) bv[nt] = load_frag_b(vr + (size_t)nt * 16 * CS, half);
#pragma unroll
        for (int nt = 0; nt < 4; ++nt)
            aco[nt] = __builtin_amdgcn_wmma_f32_16x16x32_bf16(
                false, ap0, false, bv[nt], (short)0, aco[nt], false, false);
#pragma unroll
        for (int nt = 0; nt < 4; ++nt) bv[nt] = load_frag_b(vr + (size_t)nt * 16 * CS + 32, half);
#pragma unroll
        for (int nt = 0; nt < 4; ++nt)
            aco[nt] = __builtin_amdgcn_wmma_f32_16x16x32_bf16(
                false, ap1, false, bv[nt], (short)0, aco[nt], false, false);
    }
}

// ---------------------------------------------------------------------------
// Causal flash attention. Q,K: bf16 [B,H,S,64] (Q pre-scaled); V: bf16
// [B,H,64,S]; O: bf16 [B,S,1024]. Block = 4 waves; each wave owns 16 q rows.
// Main loop is unmasked; only the diagonal block applies the causal mask.
// ---------------------------------------------------------------------------
__global__ void __launch_bounds__(128)
attn_flash(const bf16* __restrict__ Qb, const bf16* __restrict__ Kb,
           const bf16* __restrict__ Vb, bf16* __restrict__ Ob) {
    __shared__ bf16 pP[4][16 * 64];   // per-wave P tile (q-rows x keys)

    const int tid  = threadIdx.x;
    const int wave = tid >> 5;
    const int lane = tid & 31;
    const int half = lane >> 4;
    const int lr   = lane & 15;
    const int h    = blockIdx.y;
    const int b    = blockIdx.z;
    const int bh   = b * CH + h;
    const int qtile = blockIdx.x * 64;
    const int q0w   = qtile + wave * 16;

    const bf16* Kbase = Kb + (size_t)bh * CS * CDK;
    const bf16* Vbase = Vb + (size_t)bh * CDK * CS;

    v16bf aq0, aq1;
    {
        const bf16* qrow = Qb + ((size_t)bh * CS + (q0w + lr)) * CDK;
        aq0 = load_frag_a(qrow, half);
        aq1 = load_frag_a(qrow + 32, half);
    }

    v8f aco[4];
#pragma unroll
    for (int t = 0; t < 4; ++t) aco[t] = vzero();
    float mrow[8], lrow[8];
#pragma unroll
    for (int e = 0; e < 8; ++e) { mrow[e] = -1e30f; lrow[e] = 0.f; }

    bf16* pw = pP[wave];

    for (int kb = 0; kb < qtile; kb += 64)                       // fully unmasked blocks
        attn_block<false>(kb, q0w, half, lr, Kbase, Vbase, pw, aq0, aq1, aco, mrow, lrow);
    attn_block<true>(qtile, q0w, half, lr, Kbase, Vbase, pw, aq0, aq1, aco, mrow, lrow);

    // ---- normalize and store O as bf16 [B,S,1024] ----
#pragma unroll
    for (int e = 0; e < 8; ++e) {
        const float inv = 1.0f / lrow[e];
        const int q = q0w + e + 8 * half;
#pragma unroll
        for (int nt = 0; nt < 4; ++nt) {
            const float o = aco[nt][e] * inv;
            Ob[((size_t)b * CS + q) * CD + h * CDK + nt * 16 + lr] = (bf16)o;
        }
    }
}

extern "C" void kernel_launch(void* const* d_in, const int* in_sizes, int n_in,
                              void* d_out, int out_size, void* d_ws, size_t ws_size,
                              hipStream_t stream) {
    const float* query = (const float*)d_in[0];
    const float* keyin = (const float*)d_in[1];
    const float* value = (const float*)d_in[2];
    const float* Wq    = (const float*)d_in[3];
    const float* bq    = (const float*)d_in[4];
    const float* Wk    = (const float*)d_in[5];
    const float* bk    = (const float*)d_in[6];
    const float* Wv    = (const float*)d_in[7];
    const float* bv    = (const float*)d_in[8];
    const float* Wo    = (const float*)d_in[9];
    const float* bo    = (const float*)d_in[10];

    const size_t MD = (size_t)CM * CD;    // 4M elems
    const size_t DD = (size_t)CD * CD;    // 1M elems

    bf16* Xq  = (bf16*)d_ws;              // bf16 activations [M,D]
    bf16* Xk  = Xq  + MD;
    bf16* Xv  = Xk  + MD;
    bf16* Wqb = Xv  + MD;                 // bf16 weights [D,D]
    bf16* Wkb = Wqb + DD;
    bf16* Wvb = Wkb + DD;
    bf16* Wob = Wvb + DD;
    bf16* Qb  = Wob + DD;                 // [B,H,S,64] (pre-scaled by 1/8)
    bf16* Kb  = Qb  + MD;                 // [B,H,S,64]
    bf16* Vb  = Kb  + MD;                 // [B,H,64,S]
    bf16* Ob  = Vb  + MD;                 // [B,S,D]
    (void)ws_size; (void)in_sizes; (void)n_in; (void)out_size;

    // 1) one-pass fp32 -> bf16
    const int nAct8 = (int)(MD / 8), nW8 = (int)(DD / 8);
    dim3 cblk(256);
    cvt_f32_bf16<<<dim3((nAct8 + 255) / 256), cblk, 0, stream>>>(query, Xq, nAct8);
    cvt_f32_bf16<<<dim3((nAct8 + 255) / 256), cblk, 0, stream>>>(keyin, Xk, nAct8);
    cvt_f32_bf16<<<dim3((nAct8 + 255) / 256), cblk, 0, stream>>>(value, Xv, nAct8);
    cvt_f32_bf16<<<dim3((nW8 + 255) / 256),  cblk, 0, stream>>>(Wq, Wqb, nW8);
    cvt_f32_bf16<<<dim3((nW8 + 255) / 256),  cblk, 0, stream>>>(Wk, Wkb, nW8);
    cvt_f32_bf16<<<dim3((nW8 + 255) / 256),  cblk, 0, stream>>>(Wv, Wvb, nW8);
    cvt_f32_bf16<<<dim3((nW8 + 255) / 256),  cblk, 0, stream>>>(Wo, Wob, nW8);

    // 2) projections (TDM-staged bf16 WMMA GEMMs); Q pre-scaled by 1/sqrt(dk)
    dim3 blk(128);
    dim3 gproj(CM / 64, CD / 64);         // 64 x 16
    gemm_wmma<1><<<gproj, blk, 0, stream>>>(Xq, Wqb, bq, Qb, 0.125f);
    gemm_wmma<1><<<gproj, blk, 0, stream>>>(Xk, Wkb, bk, Kb, 1.0f);
    gemm_wmma<2><<<gproj, blk, 0, stream>>>(Xv, Wvb, bv, Vb, 1.0f);

    // 3) causal flash attention
    dim3 gattn(CS / 64, CH, CB);          // 32 x 16 x 2
    attn_flash<<<gattn, blk, 0, stream>>>(Qb, Kb, Vb, Ob);

    // 4) output projection -> fp32 d_out
    gemm_wmma<0><<<gproj, blk, 0, stream>>>(Ob, Wob, bo, d_out, 1.0f);
}
